// SparseGradLinear_12506944766463
// MI455X (gfx1250) — matrix operand order repaired and checked by reading, not compiled, not run on hardware
//
#include <hip/hip_runtime.h>
#include <hip/hip_bf16.h>

// out = mask(x @ U^T) @ (VT @ W)^T + bias, bf16 WMMA with f32 accumulate.
// Workspace layout (bf16): Ub[768*768] | VTb[768*768] | Wt[768*768] | Wcb[768*768]  (~4.7 MB)

typedef __bf16 bf16_t;
typedef __attribute__((ext_vector_type(16))) __bf16 v16bf;
typedef __attribute__((ext_vector_type(8)))  __bf16 v8bf;
typedef __attribute__((ext_vector_type(8)))  float  v8f;

#define DIN 768
#define MT  96            // rows per workgroup (6 row tiles) -> 342 blocks for 32768 rows
#define RT  6             // 16-row tiles per block
#define LDX 776           // padded LDS row stride (bf16 elems); 776*2B = 1552B, 16B aligned

union V16 { v16bf v; v8bf h[2]; };

// A-fragment (16x32 bf16, M x K), base -> tile origin (row 0, k0). ISA 7.12.2:
// lane m=l&15; lanes<16 hold K=kb..kb+7 & kb+16..kb+23 with kb=0, lanes>=16 kb=8.
__device__ __forceinline__ v16bf load_frag_a(const bf16_t* base, int ld, int lane) {
  const int m  = lane & 15;
  const int kb = (lane < 16) ? 0 : 8;
  const bf16_t* p = base + m * ld + kb;
  V16 u;
  u.h[0] = *(const v8bf*)(p);        // K = kb .. kb+7
  u.h[1] = *(const v8bf*)(p + 16);   // K = kb+16 .. kb+23
  return u.v;
}

// B-fragment (32x16 bf16, K x N) with B[k][n] = Wrow[n][k] (weights stored [N][K] row-major).
// lane n=l&15; lanes<16 hold K=kb..kb+7 & kb+16..kb+23 with kb=0, lanes>=16 kb=8.
__device__ __forceinline__ v16bf load_frag_b(const bf16_t* base, int ld, int lane) {
  const int n  = lane & 15;
  const int kb = (lane < 16) ? 0 : 8;
  const bf16_t* p = base + n * ld + kb;
  V16 u;
  u.h[0] = *(const v8bf*)(p);        // K = kb .. kb+7
  u.h[1] = *(const v8bf*)(p + 16);   // K = kb+16 .. kb+23
  return u.v;
}

__device__ __forceinline__ v8f wmma_bf16(v16bf a, v16bf b, v8f c) {
  return __builtin_amdgcn_wmma_f32_16x16x32_bf16(false, a, false, b, (short)0, c, false, false);
}

// ---------------- Kernel 1: precision conversion + W transpose ----------------
__global__ __launch_bounds__(256) void convert_kernel(
    const float* __restrict__ U, const float* __restrict__ VT,
    const float* __restrict__ W,
    bf16_t* __restrict__ Ub, bf16_t* __restrict__ VTb, bf16_t* __restrict__ Wt)
{
  int idx = blockIdx.x * 256 + threadIdx.x;
  if (idx < DIN * DIN) {
    Ub[idx]  = (bf16_t)U[idx];
    VTb[idx] = (bf16_t)VT[idx];
    int r = idx / DIN, c = idx - r * DIN;
    Wt[c * DIN + r] = (bf16_t)W[idx];   // W^T so GEMM B-fragments read contiguous K
  }
}

// ---------------- Kernel 2: Wc = VT @ W  (bf16 out, row-major [p][c]) ----------------
// One wave per 16x16 tile: 48*48 = 2304 tiles, 8 waves/block -> 288 blocks.
__global__ __launch_bounds__(256) void prep_gemm_kernel(
    const bf16_t* __restrict__ VTb, const bf16_t* __restrict__ Wt,
    bf16_t* __restrict__ Wcb)
{
  const int wave = blockIdx.x * 8 + (threadIdx.x >> 5);
  const int lane = threadIdx.x & 31;
  const int tr = wave / 48;   // tile row (p)
  const int tc = wave % 48;   // tile col (c)
  v8f acc = {};
  for (int k0 = 0; k0 < DIN; k0 += 32) {
    v16bf a = load_frag_a(VTb + tr * 16 * DIN + k0, DIN, lane);
    v16bf b = load_frag_b(Wt  + tc * 16 * DIN + k0, DIN, lane);
    acc = wmma_bf16(a, b, acc);
  }
  const int n  = lane & 15;
  const int rb = (lane < 16) ? 0 : 8;
#pragma unroll
  for (int v = 0; v < 8; ++v)
    Wcb[(tr * 16 + rb + v) * DIN + tc * 16 + n] = (bf16_t)acc[v];
}

// Core GEMM block: 6 row tiles x 2 col tiles, K=768, A from LDS, B from global (L2).
// Each wave owns its own disjoint pair of column tiles -> zero inter-wave B redundancy.
__device__ __forceinline__ void gemm_block(const bf16_t* aBase,           // LDS, row 0 / k 0
                                           const bf16_t* __restrict__ w0, // global tile base (col tile 0)
                                           const bf16_t* __restrict__ w1, // global tile base (col tile 1)
                                           int lane, v8f acc[RT][2])
{
#pragma unroll 2
  for (int k0 = 0; k0 < DIN; k0 += 32) {
    v16bf b0 = load_frag_b(w0 + k0, DIN, lane);
    v16bf b1 = load_frag_b(w1 + k0, DIN, lane);
#pragma unroll
    for (int r = 0; r < RT; ++r) {
      v16bf a = load_frag_a(aBase + r * 16 * LDX + k0, LDX, lane);
      acc[r][0] = wmma_bf16(a, b0, acc[r][0]);
      acc[r][1] = wmma_bf16(a, b1, acc[r][1]);
    }
  }
}

// ---------------- Kernel 3: fused  out = mask(x @ Ub^T) @ Wcb^T + bias ----------------
// 8 wave32 per block, 96 rows per block; h lives only in LDS (~291 KB of 320 KB WGP LDS).
__global__ __launch_bounds__(256) void fused_kernel(
    const float* __restrict__ x, const bf16_t* __restrict__ Ub,
    const bf16_t* __restrict__ Wcb, const float* __restrict__ bias,
    float* __restrict__ out, int nrows)
{
  extern __shared__ __align__(16) bf16_t smem[];
  bf16_t* xs = smem;               // MT * LDX  (~146 KB)
  bf16_t* hs = smem + MT * LDX;    // MT * LDX  (~146 KB)

  const int tid  = threadIdx.x;
  const int wave = tid >> 5;
  const int lane = tid & 31;
  const long rowBase = (long)blockIdx.x * MT;

  // Stage x tile: f32 -> bf16 into LDS (float4-vectorized, fully coalesced).
  const int V = DIN / 4;
  for (int i = tid; i < MT * V; i += 256) {
    int r  = i / V;
    int c4 = i - r * V;
    long gr = rowBase + r;
    bf16_t* d = &xs[r * LDX + c4 * 4];
    if (gr < nrows) {
      const float4 v = ((const float4*)(x + gr * (long)DIN))[c4];
      d[0] = (bf16_t)v.x; d[1] = (bf16_t)v.y; d[2] = (bf16_t)v.z; d[3] = (bf16_t)v.w;
    } else {
      d[0] = (bf16_t)0.f; d[1] = (bf16_t)0.f; d[2] = (bf16_t)0.f; d[3] = (bf16_t)0.f;
    }
  }
  __syncthreads();

  const int n  = lane & 15;
  const int rb = (lane < 16) ? 0 : 8;

  // ---- GEMM1: hs = mask(xs @ Ub^T), bf16 into LDS ----
  // wave owns col tiles [wave*6, wave*6+5]; 3 passes of 2 tiles.
  for (int pass = 0; pass < 3; ++pass) {
    const int c0 = wave * 6 + pass * 2;
    v8f acc[RT][2];
#pragma unroll
    for (int r = 0; r < RT; ++r) { acc[r][0] = (v8f){}; acc[r][1] = (v8f){}; }

    gemm_block(xs, Ub + (long)c0 * 16 * DIN, Ub + (long)(c0 + 1) * 16 * DIN, lane, acc);

#pragma unroll
    for (int u = 0; u < 2; ++u) {
      const int col = (c0 + u) * 16 + n;
#pragma unroll
      for (int r = 0; r < RT; ++r) {
#pragma unroll
        for (int v = 0; v < 8; ++v) {
          float f = acc[r][u][v];
          f = (__builtin_fabsf(f) > 1.0e-3f) ? f : 0.0f;   // threshold sparsify
          hs[(r * 16 + rb + v) * LDX + col] = (bf16_t)f;
        }
      }
    }
  }
  __syncthreads();

  // ---- GEMM2: out = hs @ Wcb^T + bias ----
  for (int pass = 0; pass < 3; ++pass) {
    const int c0 = wave * 6 + pass * 2;
    v8f acc[RT][2];
#pragma unroll
    for (int r = 0; r < RT; ++r) { acc[r][0] = (v8f){}; acc[r][1] = (v8f){}; }

    gemm_block(hs, Wcb + (long)c0 * 16 * DIN, Wcb + (long)(c0 + 1) * 16 * DIN, lane, acc);

#pragma unroll
    for (int u = 0; u < 2; ++u) {
      const int col = (c0 + u) * 16 + n;
      const float bv = bias[col];
#pragma unroll
      for (int r = 0; r < RT; ++r) {
#pragma unroll
        for (int v = 0; v < 8; ++v) {
          const long gr = rowBase + r * 16 + rb + v;
          if (gr < nrows)
            out[gr * (long)DIN + col] = acc[r][u][v] + bv;
        }
      }
    }
  }
}

extern "C" void kernel_launch(void* const* d_in, const int* in_sizes, int n_in,
                              void* d_out, int out_size, void* d_ws, size_t ws_size,
                              hipStream_t stream) {
  const float* x    = (const float*)d_in[0];   // [b, r, 768]
  const float* W    = (const float*)d_in[1];   // [768, 768]
  const float* bias = (const float*)d_in[2];   // [768]
  const float* U    = (const float*)d_in[3];   // [768, 768]
  const float* VT   = (const float*)d_in[4];   // [768, 768]
  float* out = (float*)d_out;

  const int N = DIN * DIN;
  bf16_t* Ub  = (bf16_t*)d_ws;
  bf16_t* VTb = Ub  + N;
  bf16_t* Wt  = VTb + N;
  bf16_t* Wcb = Wt  + N;

  const int nrows = in_sizes[0] / DIN;         // b*r = 32768

  convert_kernel<<<(N + 255) / 256, 256, 0, stream>>>(U, VT, W, Ub, VTb, Wt);
  prep_gemm_kernel<<<(48 * 48) / 8, 256, 0, stream>>>(VTb, Wt, Wcb);

  const int blocks = (nrows + MT - 1) / MT;    // 342
  const size_t smem = (size_t)2 * MT * LDX * sizeof(bf16_t);  // ~291 KB of the 320 KB WGP LDS
  fused_kernel<<<blocks, 256, smem, stream>>>(x, Ub, Wcb, bias, out, nrows);
}
